// CustomMultiBoxLoss_37495064494599
// MI455X (gfx1250) — compile-verified, matrix operand order checked by reading, not run on hardware
//
#include <hip/hip_runtime.h>
#include <hip/hip_bf16.h>
#include <cstdint>

typedef __attribute__((ext_vector_type(16))) _Float16 v16h;
typedef __attribute__((ext_vector_type(8)))  float    v8f;
typedef __attribute__((ext_vector_type(4)))  int      v4i;

#define TPB        256
#define NCLS       81
#define PMAX       8732
#define THRESH     0.5f
#define NEG_RATIO  3
#define EPS_LOG    1e-5f
#define VAR0       0.1f
#define VAR1       0.1f
#define VAR2       0.2f
#define VAR3       0.2f

#if __has_builtin(__builtin_amdgcn_global_load_async_to_lds_b128)
#define HAVE_ASYNC 1
typedef __attribute__((address_space(1))) v4i* gv4p;
typedef __attribute__((address_space(3))) v4i* lv4p;
#else
#define HAVE_ASYNC 0
#endif
#if __has_builtin(__builtin_amdgcn_s_wait_asynccnt)
#define HAVE_WAITASYNC 1
#else
#define HAVE_WAITASYNC 0
#endif

__device__ __forceinline__ float iou_corner(float g0,float g1,float g2,float g3,
                                            float q0,float q1,float q2,float q3){
  float ltx = fmaxf(g0,q0), lty = fmaxf(g1,q1);
  float rbx = fminf(g2,q2), rby = fminf(g3,q3);
  float wx = fmaxf(rbx-ltx,0.f), wy = fmaxf(rby-lty,0.f);
  float inter = wx*wy;
  float ag = (g2-g0)*(g3-g1);
  float ap = (q2-q0)*(q3-q1);
  return inter / (ag + ap - inter + 1e-10f);
}

__device__ __forceinline__ float lane_shuffle(float v, int srcLane){
  return __int_as_float(__builtin_amdgcn_ds_bpermute(srcLane << 2, __float_as_int(v)));
}

// ---------------- kernel 1: per-prior best GT (argmax over M, first-max tie) ---
__global__ void k_match_prior(const float* __restrict__ priors,
                              const float* __restrict__ tb,
                              int* __restrict__ truth, int B, int P, int M){
  int idx = blockIdx.x * blockDim.x + threadIdx.x;
  if (idx >= B * P) return;
  int b = idx / P, p = idx - b * P;
  float pcx = priors[p*4+0], pcy = priors[p*4+1], pw = priors[p*4+2], ph = priors[p*4+3];
  float q0 = pcx - pw*0.5f, q1 = pcy - ph*0.5f, q2 = pcx + pw*0.5f, q3 = pcy + ph*0.5f;
  float bestv = -1.f; int bestg = 0;
  for (int g = 0; g < M; ++g){
    const float* gp = tb + ((size_t)b * M + g) * 4;
    float v = iou_corner(gp[0], gp[1], gp[2], gp[3], q0, q1, q2, q3);
    if (v > bestv){ bestv = v; bestg = g; }
  }
  truth[idx] = (bestv >= THRESH) ? bestg : -1;
}

// ---------------- kernel 2: per-GT best prior override (block per batch) -------
__global__ void k_match_gt(const float* __restrict__ priors,
                           const float* __restrict__ tb,
                           int* __restrict__ truth, int P, int M){
  int b = blockIdx.x, tid = threadIdx.x;
  __shared__ float sv[TPB];
  __shared__ int   si[TPB];
  for (int g = 0; g < M; ++g){
    const float* gp = tb + ((size_t)b * M + g) * 4;
    float g0 = gp[0], g1 = gp[1], g2 = gp[2], g3 = gp[3];
    float bestv = -1.f; int besti = 0;
    for (int p = tid; p < P; p += TPB){
      float pcx = priors[p*4+0], pcy = priors[p*4+1], pw = priors[p*4+2], ph = priors[p*4+3];
      float v = iou_corner(g0,g1,g2,g3, pcx-pw*0.5f, pcy-ph*0.5f, pcx+pw*0.5f, pcy+ph*0.5f);
      if (v > bestv){ bestv = v; besti = p; }
    }
    sv[tid] = bestv; si[tid] = besti;
    __syncthreads();
    for (int s = TPB/2; s > 0; s >>= 1){
      if (tid < s){
        if (sv[tid+s] > sv[tid] || (sv[tid+s] == sv[tid] && si[tid+s] < si[tid])){
          sv[tid] = sv[tid+s]; si[tid] = si[tid+s];
        }
      }
      __syncthreads();
    }
    if (tid == 0) truth[(size_t)b * P + si[0]] = g;
    __syncthreads();
  }
}

// ---------------- kernel 3: localization smooth-L1, per-block partial sums -----
__global__ void k_loc(const float* __restrict__ pred_loc,
                      const float* __restrict__ priors,
                      const float* __restrict__ tb,
                      const int* __restrict__ truth,
                      float* __restrict__ locpart, int B, int P, int M){
  int idx = blockIdx.x * blockDim.x + threadIdx.x;
  float s = 0.f;
  if (idx < B * P){
    int t = truth[idx];
    if (t >= 0){
      int b = idx / P, p = idx - b * P;
      const float* gp = tb + ((size_t)b * M + t) * 4;
      float gcx = (gp[0] + gp[2]) * 0.5f, gcy = (gp[1] + gp[3]) * 0.5f;
      float gw = gp[2] - gp[0], gh = gp[3] - gp[1];
      float pcx = priors[p*4+0], pcy = priors[p*4+1], pw = priors[p*4+2], ph = priors[p*4+3];
      float enc[4];
      enc[0] = (gcx - pcx) / (pw * VAR0);
      enc[1] = (gcy - pcy) / (ph * VAR1);
      enc[2] = __logf(gw / pw + EPS_LOG) / VAR2;
      enc[3] = __logf(gh / ph + EPS_LOG) / VAR3;
      const float* pl = pred_loc + (size_t)idx * 4;
      #pragma unroll
      for (int k = 0; k < 4; ++k){
        float d = pl[k] - enc[k], ad = fabsf(d);
        s += (ad < 1.f) ? 0.5f * d * d : ad - 0.5f;
      }
    }
  }
  __shared__ float rs[TPB];
  rs[threadIdx.x] = s; __syncthreads();
  for (int st = TPB/2; st > 0; st >>= 1){
    if (threadIdx.x < st) rs[threadIdx.x] += rs[threadIdx.x + st];
    __syncthreads();
  }
  if (threadIdx.x == 0) locpart[blockIdx.x] = rs[0];
}

// ---------------- kernel 4: log-softmax CE via WMMA row-sum reduction ----------
// Wave handles 16 priors (one contiguous 16x81 f32 tile = 5184B, 16B-aligned).
// Staging: GLOBAL_LOAD_ASYNC_TO_LDS_B128 (ASYNCcnt) when available.
// Reduction: exp tile (16x96, zero-padded) x ones(96x16) via 3 chained
// V_WMMA_F32_16X16X32_F16 -> per-row sum(exp).
__global__ void k_conf_ce(const float* __restrict__ conf,
                          const int* __restrict__ truth,
                          const int* __restrict__ labels,
                          float* __restrict__ ce, int B, int P, int M){
  const int wave = threadIdx.x >> 5;
  const int lane = threadIdx.x & 31;
  const int tiles_per_b = (P + 15) >> 4;
  int tile = blockIdx.x * (TPB/32) + wave;
  bool active = tile < B * tiles_per_b;
  if (!active) tile = 0;
  int b  = tile / tiles_per_b;
  int p0 = (tile - b * tiles_per_b) << 4;

  __shared__ float smem[TPB/32][16 * NCLS];     // flat 16x81 tile per wave
  float* ts = smem[wave];

  size_t gbase = ((size_t)b * P + p0) * (size_t)NCLS;
  const float* gsrc = conf + gbase;
  int nrow  = min(16, P - p0);
  int nelem = nrow * NCLS;
  if (lane == 0) __builtin_prefetch(gsrc + nelem, 0, 1);

#if HAVE_ASYNC
  if (nelem == 16 * NCLS) {
    // 324 x b128 per-lane async transfers, fully aligned, no VGPR round-trip
    for (int i = lane; i < (16 * NCLS) / 4; i += 32) {
      gv4p g = (gv4p)(uintptr_t)(gsrc + 4 * i);
      lv4p l = (lv4p)(uint32_t)(uintptr_t)(ts + 4 * i);
      __builtin_amdgcn_global_load_async_to_lds_b128(g, l, 0, 0);
    }
#if HAVE_WAITASYNC
    __builtin_amdgcn_s_wait_asynccnt(0);
#else
    asm volatile("s_wait_asynccnt 0" ::: "memory");
#endif
  } else {
    for (int i = lane; i < nelem; i += 32) ts[i] = gsrc[i];
  }
#else
  for (int i = lane; i < nelem; i += 32) ts[i] = gsrc[i];
#endif
  __syncthreads();

  const int r = lane & 15;                 // prior row within tile
  const int h = lane >> 4;                 // K-half per WMMA A layout

  // pass 1: row max (lane pair l / l^16 covers a full row)
  float m = -3.4028235e38f;
  #pragma unroll
  for (int k = 0; k < 6; ++k){
    #pragma unroll
    for (int j = 0; j < 8; ++j){
      int c = 16*k + 8*h + j;
      if (c < NCLS) m = fmaxf(m, ts[r * NCLS + c]);
    }
  }
  m = fmaxf(m, lane_shuffle(m, lane ^ 16));

  // pass 2: exp fragments, row-sum via WMMA with B = all-ones
  v16h ones;
  #pragma unroll
  for (int i = 0; i < 16; ++i) ones[i] = (_Float16)1.0f;

  v8f acc = {};
  #pragma unroll
  for (int ch = 0; ch < 3; ++ch){
    v16h a;
    #pragma unroll
    for (int j = 0; j < 8; ++j){
      int c0 = 32*ch + 8*h + j;        // A elems 0..7  -> K = 8h + j
      int c1 = c0 + 16;                // A elems 8..15 -> K = 16 + 8h + j
      float e0 = (c0 < NCLS) ? __expf(ts[r * NCLS + c0] - m) : 0.f;
      float e1 = (c1 < NCLS) ? __expf(ts[r * NCLS + c1] - m) : 0.f;
      a[j]     = (_Float16)e0;
      a[j + 8] = (_Float16)e1;
    }
    acc = __builtin_amdgcn_wmma_f32_16x16x32_f16(false, a, false, ones,
                                                 (short)0, acc, false, false);
  }

  // D layout: VGPR j / lanes 0-15 -> rowsum(j); lanes 16-31 -> rowsum(j+8)
  int e = lane & 7;
  float x = (e==0)?acc[0]:(e==1)?acc[1]:(e==2)?acc[2]:(e==3)?acc[3]:
            (e==4)?acc[4]:(e==5)?acc[5]:(e==6)?acc[6]:acc[7];
  int src = (r < 8) ? r : (r + 8);
  float sumexp = lane_shuffle(x, src);

  int p = p0 + r;
  if (active && h == 0 && p < P){
    size_t bp = (size_t)b * P + p;
    int t   = truth[bp];
    int cls = (t >= 0) ? labels[b * M + t] : 0;
    float xt = ts[r * NCLS + cls];
    ce[bp] = m + __logf(sumexp) - xt;    // -log_softmax[target]
  }
}

// ---------------- kernel 5: hard-negative mining (exact top-k via bit bisect) --
__global__ void k_mine(const float* __restrict__ ce,
                       const int* __restrict__ truth,
                       float* __restrict__ selsum,
                       float* __restrict__ selcnt, int P){
  const int b = blockIdx.x, tid = threadIdx.x;
  __shared__ float ce_s[PMAX];
  __shared__ float rf[TPB];
  __shared__ int   ri[TPB];

  float psum = 0.f; int pcnt = 0;
  for (int i = tid; i < P; i += TPB){
    size_t bp = (size_t)b * P + i;
    float v  = ce[bp];
    bool pos = truth[bp] >= 0;
    psum += pos ? v : 0.f;
    pcnt += pos ? 1 : 0;
    ce_s[i] = pos ? 0.f : v;             // ce_neg
  }
  rf[tid] = psum; ri[tid] = pcnt; __syncthreads();
  for (int s = TPB/2; s > 0; s >>= 1){
    if (tid < s){ rf[tid] += rf[tid+s]; ri[tid] += ri[tid+s]; }
    __syncthreads();
  }
  float pos_sum = rf[0]; int pos_cnt = ri[0];
  __syncthreads();

  int neg_num = min(NEG_RATIO * pos_cnt, P - 1);

  // smallest t with count(ce_neg > t) <= neg_num; ce>=0 so float order==uint order
  unsigned lo = 0u, hi = 0x7f800000u;
  while (lo < hi){
    unsigned mid = (lo + hi) >> 1;
    float t = __uint_as_float(mid);
    int c = 0;
    for (int i = tid; i < P; i += TPB) c += (ce_s[i] > t) ? 1 : 0;
    ri[tid] = c; __syncthreads();
    for (int s = TPB/2; s > 0; s >>= 1){
      if (tid < s) ri[tid] += ri[tid+s];
      __syncthreads();
    }
    int total = ri[0];
    __syncthreads();
    if (total <= neg_num) hi = mid; else lo = mid + 1;
  }
  float tstar = __uint_as_float(lo);

  float ssum = 0.f; int cgt = 0;
  for (int i = tid; i < P; i += TPB){
    float v = ce_s[i];
    if (v > tstar){ ssum += v; cgt += 1; }
  }
  rf[tid] = ssum; ri[tid] = cgt; __syncthreads();
  for (int s = TPB/2; s > 0; s >>= 1){
    if (tid < s){ rf[tid] += rf[tid+s]; ri[tid] += ri[tid+s]; }
    __syncthreads();
  }
  if (tid == 0){
    // ties at tstar fill the remaining (neg_num - cgt) slots, each worth tstar
    selsum[b] = pos_sum + rf[0] + (float)(neg_num - ri[0]) * tstar;
    selcnt[b] = (float)(pos_cnt + neg_num);
  }
}

// ---------------- kernel 6: deterministic serial finalize ----------------------
__global__ void k_finalize(const float* __restrict__ locpart, int nlp,
                           const float* __restrict__ selsum,
                           const float* __restrict__ selcnt, int B,
                           float* __restrict__ out){
  if (blockIdx.x == 0 && threadIdx.x == 0){
    float l = 0.f;
    for (int i = 0; i < nlp; ++i) l += locpart[i];
    float n = 0.f, c = 0.f;
    for (int i = 0; i < B; ++i){ n += selsum[i]; c += selcnt[i]; }
    out[0] = l / (float)B;                       // loc_loss
    out[1] = (n / c) / (float)B;                 // conf_loss (CONF_WEIGHT = 1)
  }
}

extern "C" void kernel_launch(void* const* d_in, const int* in_sizes, int n_in,
                              void* d_out, int out_size, void* d_ws, size_t ws_size,
                              hipStream_t stream){
  const float* pred_loc  = (const float*)d_in[0];
  const float* pred_conf = (const float*)d_in[1];
  const float* priors    = (const float*)d_in[2];
  const float* tboxes    = (const float*)d_in[3];
  const int*   tlabels   = (const int*)d_in[4];

  const int P = in_sizes[2] / 4;                 // 8732
  const int B = in_sizes[0] / (P * 4);           // 64
  const int M = in_sizes[4] / B;                 // 20
  const int BP = B * P;
  const int nlp = (BP + TPB - 1) / TPB;

  int*   truth   = (int*)d_ws;
  float* ce      = (float*)d_ws + BP;
  float* locpart = ce + BP;
  float* selsum  = locpart + nlp;
  float* selcnt  = selsum + B;

  k_match_prior<<<nlp, TPB, 0, stream>>>(priors, tboxes, truth, B, P, M);
  k_match_gt<<<B, TPB, 0, stream>>>(priors, tboxes, truth, P, M);
  k_loc<<<nlp, TPB, 0, stream>>>(pred_loc, priors, tboxes, truth, locpart, B, P, M);

  int tiles   = B * ((P + 15) / 16);
  int cblocks = (tiles + (TPB/32) - 1) / (TPB/32);
  k_conf_ce<<<cblocks, TPB, 0, stream>>>(pred_conf, truth, tlabels, ce, B, P, M);

  k_mine<<<B, TPB, 0, stream>>>(ce, truth, selsum, selcnt, P);
  k_finalize<<<1, 64, 0, stream>>>(locpart, nlp, selsum, selcnt, B, (float*)d_out);
}
